// RLEModel_18571438588340
// MI455X (gfx1250) — matrix operand order repaired and checked by reading, not compiled, not run on hardware
//
#include <hip/hip_runtime.h>
#include <stdint.h>

#define B_DIM   128
#define T_DIM   512
#define IN_DIM  448
#define HID_DIM 896
#define G4_DIM  (4 * HID_DIM)
#define FEAT_DIM 16
#define KA_K    912          // HID + FEAT
#define KA_KP   928          // padded to multiple of 32 for WMMA K-loop
#define ROWS_BT (B_DIM * T_DIM)

typedef __bf16 v16bf __attribute__((ext_vector_type(16)));
typedef __bf16 v8bf  __attribute__((ext_vector_type(8)));
typedef float  v8f   __attribute__((ext_vector_type(8)));

// ---------------------------------------------------------------------------
// WMMA fragment helpers (16-bit A 16x32 / B 32x16; this code's contiguous K order, same for A and B)
//   A: lane holds row M=lane&15, K = k + (lane>>4)*16 .. +15 contiguous
//      -> one contiguous 32B run per lane (matches B's K order).
//   B: lane holds column N=lane&15, K = k + (lane>>4)*16 .. +15 contiguous.
//      Since B = W^T with W row-major [N,K], that is a contiguous 32B run of W.
// ---------------------------------------------------------------------------
__device__ __forceinline__ v16bf load_a_frag(const __bf16* rowPtr, int k, int half) {
  return *(const v16bf*)(rowPtr + k + half * 16);
}
__device__ __forceinline__ v16bf load_b_frag(const __bf16* wRowPtr, int k, int half) {
  return *(const v16bf*)(wRowPtr + k + half * 16);
}
__device__ __forceinline__ v8f wmma_bf16(v16bf a, v16bf b, v8f c) {
  // D = A x B + C, f32 accumulate  -> v_wmma_f32_16x16x32_bf16
  return __builtin_amdgcn_wmma_f32_16x16x32_bf16(false, a, false, b,
                                                 (short)0, c, false, false);
}
__device__ __forceinline__ float sigmoidf_(float x) {
  return 1.0f / (1.0f + __expf(-x));
}

// Async global -> LDS copy of 16B (CDNA5 GLOBAL_LOAD_ASYNC_TO_LDS_B128),
// tracked on ASYNCcnt. ldsAddr is the wave-relative LDS byte address (low 32
// bits of the generic shared pointer).
__device__ __forceinline__ void async_copy_b128(unsigned ldsAddr, const void* gptr) {
  asm volatile("global_load_async_to_lds_b128 %0, %1, off"
               :
               : "v"(ldsAddr), "v"((unsigned long long)(uintptr_t)gptr)
               : "memory");
}
__device__ __forceinline__ void wait_asynccnt_le2() {
  asm volatile("s_wait_asynccnt 0x2" ::: "memory");
}
__device__ __forceinline__ void wait_asynccnt_0() {
  asm volatile("s_wait_asynccnt 0x0" ::: "memory");
}

// ---------------------------------------------------------------------------
// Elementwise helpers
// ---------------------------------------------------------------------------
__global__ void cvt_pad_bf16_kernel(const float* __restrict__ src,
                                    __bf16* __restrict__ dst,
                                    long long rows, int cols, int ldd) {
  long long total = rows * (long long)ldd;
  for (long long i = blockIdx.x * (long long)blockDim.x + threadIdx.x; i < total;
       i += (long long)gridDim.x * blockDim.x) {
    long long r = i / ldd;
    int c = (int)(i - r * ldd);
    dst[i] = (c < cols) ? (__bf16)src[r * cols + c] : (__bf16)0.0f;
  }
}

__global__ void fill_zero_kernel(uint32_t* __restrict__ p, long long n) {
  for (long long i = blockIdx.x * (long long)blockDim.x + threadIdx.x; i < n;
       i += (long long)gridDim.x * blockDim.x)
    p[i] = 0u;
}

// ---------------------------------------------------------------------------
// LSTM step: wave = (mTile over B, jTile over hidden cols). Computes all 4
// gate tiles for (mTile, jTile) with shared A fragments, then fuses the
// pointwise cell update (exclusive ownership of c[row][j]). Weights are
// L2-resident; this is the latency-critical chain, so no LDS staging here.
// ---------------------------------------------------------------------------
__global__ __launch_bounds__(32) void lstm_step_kernel(
    const __bf16* __restrict__ x_bf,    // [B, T, IN]
    const __bf16* __restrict__ Wih,     // [4H, IN]
    const __bf16* __restrict__ Whh,     // [4H, H]
    const float* __restrict__ b_ih,     // [4H]
    const float* __restrict__ b_hh,     // [4H]
    const __bf16* __restrict__ h_in,    // [B, H]
    __bf16* __restrict__ h_out,         // [B, H]
    float* __restrict__ c_buf,          // [B, H]
    float* __restrict__ h_f32,          // [B, H]
    __bf16* __restrict__ h_hist,        // [B, T, H]
    int t) {
  const int lane = threadIdx.x & 31;
  const int half = lane >> 4;
  const int l15 = lane & 15;
  const int jT = blockIdx.x;            // 0..55
  const int mT = blockIdx.y;            // 0..7

  const int aRow = mT * 16 + l15;
  const __bf16* xRow = x_bf + ((size_t)aRow * T_DIM + t) * IN_DIM;
  const __bf16* hRow = h_in + (size_t)aRow * HID_DIM;

  const int j = jT * 16 + l15;          // this lane's B column (gate hidden idx)
  const __bf16* wiRow[4];
  const __bf16* whRow[4];
#pragma unroll
  for (int q = 0; q < 4; ++q) {
    const int o = q * HID_DIM + j;
    wiRow[q] = Wih + (size_t)o * IN_DIM;
    whRow[q] = Whh + (size_t)o * HID_DIM;
  }

  v8f acc[4] = {v8f{}, v8f{}, v8f{}, v8f{}};

  // x_t @ W_ih^T
  for (int k = 0; k < IN_DIM; k += 32) {
    const v16bf a = load_a_frag(xRow, k, half);
#pragma unroll
    for (int q = 0; q < 4; ++q)
      acc[q] = wmma_bf16(a, load_b_frag(wiRow[q], k, half), acc[q]);
  }
  // h_{t-1} @ W_hh^T
  for (int k = 0; k < HID_DIM; k += 32) {
    const v16bf a = load_a_frag(hRow, k, half);
#pragma unroll
    for (int q = 0; q < 4; ++q)
      acc[q] = wmma_bf16(a, load_b_frag(whRow[q], k, half), acc[q]);
  }

  float bsum[4];
#pragma unroll
  for (int q = 0; q < 4; ++q)
    bsum[q] = b_ih[q * HID_DIM + j] + b_hh[q * HID_DIM + j];

#pragma unroll
  for (int r = 0; r < 8; ++r) {
    const int row = mT * 16 + half * 8 + r;      // C layout: vgpr r, lane half
    const float gi = acc[0][r] + bsum[0];
    const float gf = acc[1][r] + bsum[1];
    const float gg = acc[2][r] + bsum[2];
    const float go = acc[3][r] + bsum[3];
    const float iv = sigmoidf_(gi);
    const float fv = sigmoidf_(gf);
    const float gv = tanhf(gg);
    const float ov = sigmoidf_(go);
    const size_t idx = (size_t)row * HID_DIM + j;
    const float c = fv * c_buf[idx] + iv * gv;
    c_buf[idx] = c;
    const float h = ov * tanhf(c);
    h_f32[idx] = h;
    const __bf16 hb = (__bf16)h;
    h_out[idx] = hb;
    h_hist[((size_t)row * T_DIM + t) * HID_DIM + j] = hb;
  }
}

// ---------------------------------------------------------------------------
// Simple per-wave WMMA GEMM (used for the N=16 output layer only).
// ---------------------------------------------------------------------------
__global__ __launch_bounds__(32) void gemm_bf16_kernel(
    const __bf16* __restrict__ A, int lda,
    const __bf16* __restrict__ W, int ldw,
    const float* __restrict__ bias,
    __bf16* __restrict__ Cb,       // optional bf16 out (row stride = N)
    float* __restrict__ Cf,        // optional f32 out  (row stride = N)
    int N, int K, int relu) {
  const int lane = threadIdx.x & 31;
  const int half = lane >> 4;
  const int l15 = lane & 15;
  const int nT = blockIdx.x;
  const int mT = blockIdx.y;

  const __bf16* aRow = A + (size_t)(mT * 16 + l15) * lda;
  const int n = nT * 16 + l15;
  const __bf16* wRow = W + (size_t)n * ldw;

  v8f acc = v8f{};
  for (int k = 0; k < K; k += 32) {
    __builtin_prefetch(wRow + k + 256, 0, 1);   // global_prefetch_b8
    __builtin_prefetch(aRow + k + 256, 0, 1);
    acc = wmma_bf16(load_a_frag(aRow, k, half), load_b_frag(wRow, k, half), acc);
  }

  const float bv = bias ? bias[n] : 0.0f;
#pragma unroll
  for (int r = 0; r < 8; ++r) {
    const int row = mT * 16 + half * 8 + r;
    float v = acc[r] + bv;
    if (relu) v = fmaxf(v, 0.0f);
    const size_t idx = (size_t)row * N + n;
    if (Cf) Cf[idx] = v;
    if (Cb) Cb[idx] = (__bf16)v;
  }
}

// ---------------------------------------------------------------------------
// LDS-staged WMMA GEMM: block = 4 waves computing a 64(M) x 64(N) macro-tile.
// The shared 64x32 bf16 weight tile (4 KB) is streamed into LDS with
// GLOBAL_LOAD_ASYNC_TO_LDS_B128 (ASYNCcnt), double-buffered so the next K
// chunk overlaps the current 16 WMMAs. Requires M%64==0, N%64==0, K%32==0.
// Per K-step: 8 KB global traffic for 256 KFLOP -> 32 FLOP/byte.
// ---------------------------------------------------------------------------
__global__ __launch_bounds__(128) void gemm_bf16_lds_kernel(
    const __bf16* __restrict__ A, int lda,
    const __bf16* __restrict__ W, int ldw,
    const float* __restrict__ bias,
    __bf16* __restrict__ Cb,       // optional bf16 out (row stride = N)
    float* __restrict__ Cf,        // optional f32 out  (row stride = N)
    int N, int K, int relu) {
  __shared__ __align__(64) __bf16 stage[2][64 * 32];   // 2 x 4 KB

  const int tid = threadIdx.x;
  const int wave = tid >> 5;
  const int lane = tid & 31;
  const int half = lane >> 4;
  const int l15 = lane & 15;
  const int nBase = blockIdx.x * 64;
  const int mBase = blockIdx.y * 64 + wave * 16;

  const __bf16* aRow = A + (size_t)(mBase + l15) * lda;
  const unsigned sbase = (unsigned)(uintptr_t)(&stage[0][0]);

  // Each thread owns two 16B pieces of the 4 KB weight tile:
  //   piece p (0..255): row = p>>2 (0..63), quarter q = p&3.
  const int p0 = tid, p1 = tid + 128;
  const int r0 = p0 >> 2, q0 = p0 & 3;
  const int r1 = p1 >> 2, q1 = p1 & 3;
  const __bf16* w0 = W + (size_t)(nBase + r0) * ldw + q0 * 8;
  const __bf16* w1 = W + (size_t)(nBase + r1) * ldw + q1 * 8;
  const unsigned s0 = (unsigned)(r0 * 64 + q0 * 16);
  const unsigned s1 = (unsigned)(r1 * 64 + q1 * 16);

  v8f acc[4] = {v8f{}, v8f{}, v8f{}, v8f{}};
  const int KT = K / 32;

  // Prologue: fill stage 0 with K-chunk 0.
  async_copy_b128(sbase + s0, w0);
  async_copy_b128(sbase + s1, w1);

  for (int kt = 0; kt < KT; ++kt) {
    const int cur = kt & 1;
    const int k = kt * 32;
    if (kt + 1 < KT) {
      const unsigned nb = sbase + (unsigned)((cur ^ 1) * 64 * 32 * 2);
      async_copy_b128(nb + s0, w0 + (kt + 1) * 32);
      async_copy_b128(nb + s1, w1 + (kt + 1) * 32);
      wait_asynccnt_le2();      // in-order: current chunk's 2 copies landed
    } else {
      wait_asynccnt_0();
    }
    __syncthreads();            // everyone's slice of the tile is visible

    const v16bf a = load_a_frag(aRow, k, half);
    const __bf16* st = &stage[cur][0];
#pragma unroll
    for (int nt = 0; nt < 4; ++nt) {
      const v16bf b = *(const v16bf*)(st + (nt * 16 + l15) * 32 + half * 16);
      acc[nt] = wmma_bf16(a, b, acc[nt]);
    }
    __syncthreads();            // protect buffer reused two iterations later
  }

#pragma unroll
  for (int nt = 0; nt < 4; ++nt) {
    const int n = nBase + nt * 16 + l15;
    const float bv = bias ? bias[n] : 0.0f;
#pragma unroll
    for (int r = 0; r < 8; ++r) {
      const int row = mBase + half * 8 + r;
      float v = acc[nt][r] + bv;
      if (relu) v = fmaxf(v, 0.0f);
      const size_t idx = (size_t)row * N + n;
      if (Cf) Cf[idx] = v;
      if (Cb) Cb[idx] = (__bf16)v;
    }
  }
}

// ---------------------------------------------------------------------------
// mu normalize: thread per row of 16; writes f32 out + bf16 copy for concat
// ---------------------------------------------------------------------------
__global__ void mu_norm_kernel(const float* __restrict__ mu_raw,
                               float* __restrict__ mu_out,
                               __bf16* __restrict__ mu_bf, int rows) {
  const int r = blockIdx.x * blockDim.x + threadIdx.x;
  if (r >= rows) return;
  float v[FEAT_DIM];
  float s = 0.0f;
#pragma unroll
  for (int i = 0; i < FEAT_DIM; ++i) {
    v[i] = mu_raw[(size_t)r * FEAT_DIM + i];
    s += v[i] * v[i];
  }
  const float nrm = fmaxf(sqrtf(s), 1e-12f);
#pragma unroll
  for (int i = 0; i < FEAT_DIM; ++i) {
    const float o = v[i] / nrm;
    mu_out[(size_t)r * FEAT_DIM + i] = o;
    mu_bf[(size_t)r * FEAT_DIM + i] = (__bf16)o;
  }
}

// comb[row, 0:896] = h_hist[row], comb[row, 896:912] = mu_bf[row], pad 0
__global__ void concat_kernel(const __bf16* __restrict__ h_hist,
                              const __bf16* __restrict__ mu_bf,
                              __bf16* __restrict__ comb, long long rows) {
  const long long total = rows * (long long)KA_KP;
  for (long long i = blockIdx.x * (long long)blockDim.x + threadIdx.x; i < total;
       i += (long long)gridDim.x * blockDim.x) {
    const long long row = i / KA_KP;
    const int c = (int)(i - row * KA_KP);
    __bf16 v = (__bf16)0.0f;
    if (c < HID_DIM) v = h_hist[row * HID_DIM + c];
    else if (c < KA_K) v = mu_bf[row * FEAT_DIM + (c - HID_DIM)];
    comb[i] = v;
  }
}

// kappa layer3: N=1 dot (K=448) with wave reduction, then softplus + clip
__global__ __launch_bounds__(256) void kappa3_kernel(
    const __bf16* __restrict__ a2,     // [rows, 448]
    const float* __restrict__ W3,      // [448] (fp32 original)
    const float* __restrict__ b3,      // [1]
    float* __restrict__ kout, int rows) {
  const int wave = (blockIdx.x * blockDim.x + threadIdx.x) >> 5;
  const int lane = threadIdx.x & 31;
  if (wave >= rows) return;
  float s = 0.0f;
  for (int k = lane; k < HID_DIM / 2; k += 32)
    s += (float)a2[(size_t)wave * (HID_DIM / 2) + k] * W3[k];
#pragma unroll
  for (int off = 16; off > 0; off >>= 1) s += __shfl_xor(s, off, 32);
  if (lane == 0) {
    const float x = s + b3[0];
    const float sp = (x > 20.0f) ? x : log1pf(__expf(x));
    kout[wave] = fminf(fmaxf(sp, 0.1f), 50.0f);
  }
}

__global__ void copy2_kernel(const float* __restrict__ h, const float* __restrict__ c,
                             float* __restrict__ hN, float* __restrict__ cN, int n) {
  const int i = blockIdx.x * blockDim.x + threadIdx.x;
  if (i < n) { hN[i] = h[i]; cN[i] = c[i]; }
}

// ---------------------------------------------------------------------------
// Host-side launch
// ---------------------------------------------------------------------------
extern "C" void kernel_launch(void* const* d_in, const int* in_sizes, int n_in,
                              void* d_out, int out_size, void* d_ws, size_t ws_size,
                              hipStream_t stream) {
  (void)in_sizes; (void)n_in; (void)out_size; (void)ws_size;

  const float* states = (const float*)d_in[0];
  const float* W_ih   = (const float*)d_in[1];
  const float* W_hh   = (const float*)d_in[2];
  const float* b_ih   = (const float*)d_in[3];
  const float* b_hh   = (const float*)d_in[4];
  const float* mu_W1  = (const float*)d_in[5];
  const float* mu_b1  = (const float*)d_in[6];
  const float* mu_W2  = (const float*)d_in[7];
  const float* mu_b2  = (const float*)d_in[8];
  const float* mu_W3  = (const float*)d_in[9];
  const float* mu_b3  = (const float*)d_in[10];
  const float* ka_W1  = (const float*)d_in[11];
  const float* ka_b1  = (const float*)d_in[12];
  const float* ka_W2  = (const float*)d_in[13];
  const float* ka_b2  = (const float*)d_in[14];
  const float* ka_W3  = (const float*)d_in[15];
  const float* ka_b3  = (const float*)d_in[16];

  float* out = (float*)d_out;
  float* mu_out    = out;                                     // [BT, 16]
  float* kappa_out = out + (size_t)ROWS_BT * FEAT_DIM;        // [BT, 1]
  float* hN_out    = kappa_out + ROWS_BT;                     // [B, H]
  float* cN_out    = hN_out + (size_t)B_DIM * HID_DIM;        // [B, H]

  // ---- workspace bump allocator (256B aligned) ----
  char* ws = (char*)d_ws;
  size_t off = 0;
  auto alloc = [&](size_t bytes) -> void* {
    off = (off + 255) & ~(size_t)255;
    void* p = ws + off;
    off += bytes;
    return p;
  };
  __bf16* x_bf    = (__bf16*)alloc((size_t)ROWS_BT * IN_DIM * 2);
  __bf16* Wih_bf  = (__bf16*)alloc((size_t)G4_DIM * IN_DIM * 2);
  __bf16* Whh_bf  = (__bf16*)alloc((size_t)G4_DIM * HID_DIM * 2);
  __bf16* muW1_bf = (__bf16*)alloc((size_t)HID_DIM * HID_DIM * 2);
  __bf16* muW2_bf = (__bf16*)alloc((size_t)(HID_DIM / 2) * HID_DIM * 2);
  __bf16* muW3_bf = (__bf16*)alloc((size_t)FEAT_DIM * (HID_DIM / 2) * 2);
  __bf16* kaW1_bf = (__bf16*)alloc((size_t)HID_DIM * KA_KP * 2);
  __bf16* kaW2_bf = (__bf16*)alloc((size_t)(HID_DIM / 2) * HID_DIM * 2);
  __bf16* h_ping  = (__bf16*)alloc((size_t)B_DIM * HID_DIM * 2);
  __bf16* h_pong  = (__bf16*)alloc((size_t)B_DIM * HID_DIM * 2);
  float*  c_buf   = (float*)alloc((size_t)B_DIM * HID_DIM * 4);
  float*  h_f32   = (float*)alloc((size_t)B_DIM * HID_DIM * 4);
  __bf16* h_hist  = (__bf16*)alloc((size_t)ROWS_BT * HID_DIM * 2);
  __bf16* a1_bf   = (__bf16*)alloc((size_t)ROWS_BT * HID_DIM * 2);
  __bf16* a2_bf   = (__bf16*)alloc((size_t)ROWS_BT * (HID_DIM / 2) * 2);
  float*  mu_raw  = (float*)alloc((size_t)ROWS_BT * FEAT_DIM * 4);
  __bf16* mu_bf   = (__bf16*)alloc((size_t)ROWS_BT * FEAT_DIM * 2);
  __bf16* comb_bf = (__bf16*)alloc((size_t)ROWS_BT * KA_KP * 2);

  const int TPB = 256;
  auto gsz = [](long long n, int tpb) {
    long long b = (n + tpb - 1) / tpb;
    return (unsigned)(b > 65536 ? 65536 : b);
  };

  // ---- 1) fp32 -> bf16 conversions (kappa W1 K-padded 912 -> 928) ----
  cvt_pad_bf16_kernel<<<gsz((long long)ROWS_BT * IN_DIM, TPB), TPB, 0, stream>>>(
      states, x_bf, ROWS_BT, IN_DIM, IN_DIM);
  cvt_pad_bf16_kernel<<<gsz((long long)G4_DIM * IN_DIM, TPB), TPB, 0, stream>>>(
      W_ih, Wih_bf, G4_DIM, IN_DIM, IN_DIM);
  cvt_pad_bf16_kernel<<<gsz((long long)G4_DIM * HID_DIM, TPB), TPB, 0, stream>>>(
      W_hh, Whh_bf, G4_DIM, HID_DIM, HID_DIM);
  cvt_pad_bf16_kernel<<<gsz((long long)HID_DIM * HID_DIM, TPB), TPB, 0, stream>>>(
      mu_W1, muW1_bf, HID_DIM, HID_DIM, HID_DIM);
  cvt_pad_bf16_kernel<<<gsz((long long)(HID_DIM / 2) * HID_DIM, TPB), TPB, 0, stream>>>(
      mu_W2, muW2_bf, HID_DIM / 2, HID_DIM, HID_DIM);
  cvt_pad_bf16_kernel<<<gsz((long long)FEAT_DIM * (HID_DIM / 2), TPB), TPB, 0, stream>>>(
      mu_W3, muW3_bf, FEAT_DIM, HID_DIM / 2, HID_DIM / 2);
  cvt_pad_bf16_kernel<<<gsz((long long)HID_DIM * KA_KP, TPB), TPB, 0, stream>>>(
      ka_W1, kaW1_bf, HID_DIM, KA_K, KA_KP);
  cvt_pad_bf16_kernel<<<gsz((long long)(HID_DIM / 2) * HID_DIM, TPB), TPB, 0, stream>>>(
      ka_W2, kaW2_bf, HID_DIM / 2, HID_DIM, HID_DIM);

  // ---- 2) zero h0, c0 ----
  fill_zero_kernel<<<gsz((long long)B_DIM * HID_DIM / 2, TPB), TPB, 0, stream>>>(
      (uint32_t*)h_ping, (long long)B_DIM * HID_DIM / 2);
  fill_zero_kernel<<<gsz((long long)B_DIM * HID_DIM, TPB), TPB, 0, stream>>>(
      (uint32_t*)c_buf, (long long)B_DIM * HID_DIM);

  // ---- 3) LSTM recurrence (sequential over T) ----
  const dim3 lstmGrid(HID_DIM / 16, B_DIM / 16);   // 56 x 8 waves
  for (int t = 0; t < T_DIM; ++t) {
    __bf16* hin = (t & 1) ? h_pong : h_ping;
    __bf16* hout = (t & 1) ? h_ping : h_pong;
    lstm_step_kernel<<<lstmGrid, 32, 0, stream>>>(
        x_bf, Wih_bf, Whh_bf, b_ih, b_hh, hin, hout, c_buf, h_f32, h_hist, t);
  }

  // ---- 4) mu_net: 896 -> 896 -> 448 -> 16 ----
  gemm_bf16_lds_kernel<<<dim3(HID_DIM / 64, ROWS_BT / 64), 128, 0, stream>>>(
      h_hist, HID_DIM, muW1_bf, HID_DIM, mu_b1, a1_bf, nullptr,
      HID_DIM, HID_DIM, 1);
  gemm_bf16_lds_kernel<<<dim3(HID_DIM / 128, ROWS_BT / 64), 128, 0, stream>>>(
      a1_bf, HID_DIM, muW2_bf, HID_DIM, mu_b2, a2_bf, nullptr,
      HID_DIM / 2, HID_DIM, 1);
  gemm_bf16_kernel<<<dim3(FEAT_DIM / 16, ROWS_BT / 16), 32, 0, stream>>>(
      a2_bf, HID_DIM / 2, muW3_bf, HID_DIM / 2, mu_b3, nullptr, mu_raw,
      FEAT_DIM, HID_DIM / 2, 0);
  mu_norm_kernel<<<(ROWS_BT + TPB - 1) / TPB, TPB, 0, stream>>>(
      mu_raw, mu_out, mu_bf, ROWS_BT);

  // ---- 5) kappa_net: concat(h, mu) 912(->928) -> 896 -> 448 -> 1 ----
  concat_kernel<<<gsz((long long)ROWS_BT * KA_KP, TPB), TPB, 0, stream>>>(
      h_hist, mu_bf, comb_bf, ROWS_BT);
  gemm_bf16_lds_kernel<<<dim3(HID_DIM / 64, ROWS_BT / 64), 128, 0, stream>>>(
      comb_bf, KA_KP, kaW1_bf, KA_KP, ka_b1, a1_bf, nullptr,
      HID_DIM, KA_KP, 1);
  gemm_bf16_lds_kernel<<<dim3(HID_DIM / 128, ROWS_BT / 64), 128, 0, stream>>>(
      a1_bf, HID_DIM, kaW2_bf, HID_DIM, ka_b2, a2_bf, nullptr,
      HID_DIM / 2, HID_DIM, 1);
  kappa3_kernel<<<(ROWS_BT * 32 + TPB - 1) / TPB, TPB, 0, stream>>>(
      a2_bf, ka_W3, ka_b3, kappa_out, ROWS_BT);

  // ---- 6) hN, cN ----
  copy2_kernel<<<(B_DIM * HID_DIM + TPB - 1) / TPB, TPB, 0, stream>>>(
      h_f32, c_buf, hN_out, cN_out, B_DIM * HID_DIM);
}
